// ReactivityNet_83674552860923
// MI455X (gfx1250) — matrix operand order repaired and checked by reading, not compiled, not run on hardware
//
#include <hip/hip_runtime.h>

#define DEV __device__ __forceinline__

typedef __attribute__((ext_vector_type(8)))  _Float16 v8h;
typedef __attribute__((ext_vector_type(16))) _Float16 v16h;
typedef __attribute__((ext_vector_type(8)))  float    v8f;

// ---- problem constants ----
constexpr int B_    = 16;
constexpr int N_    = 100;
constexpr int MAXNB_= 10;
constexpr int NBONDS_ = 120;
constexpr int AF_   = 82;
constexpr int BF_   = 6;
constexpr int H_    = 300;
constexpr int BIN_  = 11;
constexpr int K_    = 80;

constexpr int HP   = 320;   // H padded to mult of 32 (and 64 for N-tiling)
constexpr int AFP  = 96;    // AF padded
constexpr int SKP  = 32;    // small-K pad (BF=6, BIN=11)
constexpr int U1KP = 640;   // 2H padded
constexpr int ROWS  = B_ * N_;            // 1600
constexpr int GROWS = B_ * N_ * MAXNB_;   // 16000
constexpr int NJT   = (N_ + 15) / 16;     // 7 j-tiles
constexpr int NPAIR = B_ * N_ * N_;       // 160000

// ---------------------------------------------------------------------------
// WMMA fragment helpers (wave32; C/D per CDNA5 ISA 7.12.2, A/B in this code's contiguous k order)
// A 16x32 f16: lane -> row=lane&15, 16 contiguous K-halves at
//              koff=(lane>>4)*16 (matches B)
// B 32x16 f16 from transposed weights Wt[N][Kp]: lane -> col N=lane&15,
//              16 contiguous K-halves at koff=(lane>>4)*16
// C/D v8f: VGPR r -> M = r + 8*(lane>=16), N = lane&15
// ---------------------------------------------------------------------------
DEV v16h frag_a(const _Float16* base /* row ptr + asel */, int kb) {
  v8h lo = *(const v8h*)(base + kb);
  v8h hi = *(const v8h*)(base + kb + 8);
  v16h r;
#pragma unroll
  for (int t = 0; t < 8; ++t) { r[t] = lo[t]; r[t + 8] = hi[t]; }
  return r;
}

DEV v8f wmma_step(v16h a, v16h b, v8f c) {
  return __builtin_amdgcn_wmma_f32_16x16x32_f16(false, a, false, b,
                                                (short)0, c, false, false);
}

// ---------------------------------------------------------------------------
// Generic GEMM: C[M][Np] = act( A[M][Kp] * Wt(N-major)[Np][Kp] + bias + Cin )
// grid = (M/16, Np/64), block = 128 (4 waves, one 16x16 tile per wave)
// ---------------------------------------------------------------------------
__global__ void k_gemm(const _Float16* __restrict__ A,
                       const _Float16* __restrict__ Wt,
                       const float* __restrict__ bias,
                       const float* __restrict__ Cin,
                       float* __restrict__ Cf, _Float16* __restrict__ Ch,
                       int Np, int Kp, int relu)
{
  const int lane = threadIdx.x & 31;
  const int wave = threadIdx.x >> 5;
  const int m0 = blockIdx.x * 16;
  const int n0 = (blockIdx.y * 4 + wave) * 16;
  const int arow = lane & 15, asel = (lane >> 4) * 16;
  const int bn   = lane & 15, bk   = (lane >> 4) * 16;
  const int cn   = lane & 15, cm   = (lane >> 4) * 8;

  v8f acc = {};
  if (Cin) {
#pragma unroll
    for (int r = 0; r < 8; ++r)
      acc[r] = Cin[(size_t)(m0 + cm + r) * Np + n0 + cn];
  }

  const _Float16* Ap = A  + (size_t)(m0 + arow) * Kp + asel;
  const _Float16* Bp = Wt + (size_t)(n0 + bn)   * Kp + bk;
  for (int kb = 0; kb < Kp; kb += 32)
    acc = wmma_step(frag_a(Ap, kb), *(const v16h*)(Bp + kb), acc);

  const float bv = bias ? bias[n0 + cn] : 0.f;
#pragma unroll
  for (int r = 0; r < 8; ++r) {
    float v = acc[r] + bv;
    if (relu) v = v > 0.f ? v : 0.f;
    size_t o = (size_t)(m0 + cm + r) * Np + n0 + cn;
    if (Cf) Cf[o] = v;
    if (Ch) Ch[o] = (_Float16)v;
  }
}

// ---------------------------------------------------------------------------
// Packing / gather kernels
// ---------------------------------------------------------------------------
__global__ void k_pack_wt(const float* __restrict__ W, _Float16* __restrict__ Wt,
                          int K, int N, int Kp, int Np, int ldW, int row0)
{
  int idx = blockIdx.x * blockDim.x + threadIdx.x;
  if (idx >= Np * Kp) return;
  int n = idx / Kp, k = idx % Kp;
  Wt[idx] = (k < K && n < N) ? (_Float16)W[(size_t)(row0 + k) * ldW + n]
                             : (_Float16)0.f;
}

__global__ void k_pack_vec(const float* __restrict__ v, float* __restrict__ vp,
                           int N, int Np)
{
  int idx = blockIdx.x * blockDim.x + threadIdx.x;
  if (idx >= Np) return;
  vp[idx] = (idx < N) ? v[idx] : 0.f;
}

__global__ void k_pack_fatoms(const float* __restrict__ X, _Float16* __restrict__ A)
{
  int idx = blockIdx.x * blockDim.x + threadIdx.x;
  if (idx >= ROWS * AFP) return;
  int r = idx / AFP, k = idx % AFP;
  A[idx] = (k < AF_) ? (_Float16)X[(size_t)r * AF_ + k] : (_Float16)0.f;
}

__global__ void k_gather_nei(const _Float16* __restrict__ h16,
                             const int* __restrict__ atom_nb,
                             _Float16* __restrict__ A)
{
  int idx = blockIdx.x * blockDim.x + threadIdx.x;
  if (idx >= GROWS * HP) return;
  int g = idx / HP, k = idx % HP;
  int b = g / (N_ * MAXNB_);
  int src = b * N_ + atom_nb[g];
  A[idx] = h16[(size_t)src * HP + k];
}

__global__ void k_pack_bond(const float* __restrict__ fbonds,
                            const int* __restrict__ bond_nb,
                            _Float16* __restrict__ A)
{
  int idx = blockIdx.x * blockDim.x + threadIdx.x;
  if (idx >= GROWS * SKP) return;
  int g = idx / SKP, k = idx % SKP;
  int b = g / (N_ * MAXNB_);
  A[idx] = (k < BF_)
    ? (_Float16)fbonds[((size_t)(b * NBONDS_ + bond_nb[g])) * BF_ + k]
    : (_Float16)0.f;
}

__global__ void k_pack_u1(const _Float16* __restrict__ h16,
                          const float* __restrict__ nei_label,
                          _Float16* __restrict__ A)
{
  int idx = blockIdx.x * blockDim.x + threadIdx.x;
  if (idx >= ROWS * U1KP) return;
  int r = idx / U1KP, k = idx % U1KP;
  _Float16 v = (_Float16)0.f;
  if (k < H_)           v = h16[(size_t)r * HP + k];
  else if (k < 2 * H_)  v = (_Float16)nei_label[(size_t)r * HP + (k - H_)];
  A[idx] = v;
}

// ---------------------------------------------------------------------------
// Elementwise / reduction kernels (grid = ROWS blocks, 320 threads)
// ---------------------------------------------------------------------------
__global__ void k_fnei(const float* __restrict__ Hnei, const float* __restrict__ Bnei,
                       const float* __restrict__ mask_nei, float* __restrict__ fnei)
{
  int bi = blockIdx.x, h = threadIdx.x;
  float s = 0.f;
#pragma unroll
  for (int m = 0; m < MAXNB_; ++m) {
    size_t g = (size_t)bi * MAXNB_ + m;
    s += Hnei[g * HP + h] * Bnei[g * HP + h] * mask_nei[g];
  }
  fnei[(size_t)bi * HP + h] = s;
}

__global__ void k_local(const float* __restrict__ fnei, const float* __restrict__ fself,
                        const float* __restrict__ mask_at, float* __restrict__ localf)
{
  int bi = blockIdx.x, h = threadIdx.x;
  size_t o = (size_t)bi * HP + h;
  localf[o] = fnei[o] * fself[o] * mask_at[bi];
}

__global__ void k_labelsum(const float* __restrict__ U2out,
                           const float* __restrict__ mask_nei,
                           float* __restrict__ nei_label)
{
  int bi = blockIdx.x, h = threadIdx.x;
  float s = 0.f;
#pragma unroll
  for (int m = 0; m < MAXNB_; ++m) {
    size_t g = (size_t)bi * MAXNB_ + m;
    s += U2out[g * HP + h] * mask_nei[g];
  }
  nei_label[(size_t)bi * HP + h] = s;
}

__global__ void k_ctx(const float* __restrict__ att_s, const float* __restrict__ localf,
                      float* __restrict__ ctx)
{
  int bi = blockIdx.x, h = threadIdx.x;
  int b = bi / N_;
  float s = 0.f;
  for (int j = 0; j < N_; ++j)
    s += att_s[(size_t)bi * N_ + j] * localf[((size_t)(b * N_ + j)) * HP + h];
  ctx[(size_t)bi * HP + h] = s;
}

// ---------------------------------------------------------------------------
// Fused pair-attention: att_s[b,i,j] =
//   sigmoid( relu((local_i+local_j)@Wl + bin@Wb + b_att) . wscore + b_score )
// one block = (b, i, 16 j's); 4 waves x 5 N-tiles of the H=320 hidden dim
// ---------------------------------------------------------------------------
__global__ void k_att(const float* __restrict__ localf, const float* __restrict__ binf,
                      const _Float16* __restrict__ Wtl, const _Float16* __restrict__ Wtb,
                      const float* __restrict__ biasp, const float* __restrict__ wscore,
                      const float* __restrict__ bscore, float* __restrict__ att_s)
{
  __shared__ __align__(32) _Float16 sA[16 * HP];
  __shared__ __align__(32) _Float16 sBin[16 * SKP];
  __shared__ float sdot[16];

  int blk = blockIdx.x;
  int jt = blk % NJT; int t = blk / NJT;
  int i = t % N_; int b = t / N_;
  int j0 = jt * 16;
  int tid = threadIdx.x;

  const float* li = localf + ((size_t)(b * N_ + i)) * HP;
  for (int idx = tid; idx < 16 * HP; idx += 128) {
    int jj = idx / HP, k = idx % HP;
    int j = j0 + jj; int jc = j < N_ ? j : N_ - 1;
    sA[idx] = (_Float16)(li[k] + localf[((size_t)(b * N_ + jc)) * HP + k]);
  }
  for (int idx = tid; idx < 16 * SKP; idx += 128) {
    int jj = idx / SKP, k = idx % SKP;
    int j = j0 + jj; int jc = j < N_ ? j : N_ - 1;
    sBin[idx] = (k < BIN_)
      ? (_Float16)binf[((size_t)((b * N_ + i) * N_ + jc)) * BIN_ + k]
      : (_Float16)0.f;
  }
  if (tid < 16) sdot[tid] = bscore[0];
  __syncthreads();

  const int lane = tid & 31, wave = tid >> 5;
  const int arow = lane & 15, asel = (lane >> 4) * 16;
  const int bn   = lane & 15, bk   = (lane >> 4) * 16;
  const int cn   = lane & 15, cm   = (lane >> 4) * 8;

  for (int nt = wave; nt < HP / 16; nt += 4) {
    int n0 = nt * 16;
    v8f acc = {};
    const _Float16* Ap = sA + arow * HP + asel;
    const _Float16* Bp = Wtl + (size_t)(n0 + bn) * HP + bk;
    for (int kb = 0; kb < HP; kb += 32)
      acc = wmma_step(frag_a(Ap, kb), *(const v16h*)(Bp + kb), acc);
    acc = wmma_step(frag_a(sBin + arow * SKP + asel, 0),
                    *(const v16h*)(Wtb + (size_t)(n0 + bn) * SKP + bk), acc);
    float wsn = wscore[n0 + cn];
    float bvn = biasp[n0 + cn];
#pragma unroll
    for (int r = 0; r < 8; ++r) {
      float v = acc[r] + bvn;
      v = v > 0.f ? v : 0.f;
      atomicAdd(&sdot[cm + r], v * wsn);
    }
  }
  __syncthreads();
  if (tid < 16) {
    int j = j0 + tid;
    if (j < N_) {
      float s = 1.f / (1.f + __expf(-sdot[tid]));
      att_s[(size_t)(b * N_ + i) * N_ + j] = s;
    }
  }
}

// ---------------------------------------------------------------------------
// Fused final head: score[b,i,j] =
//   relu(lp@Wpl + gp@Wpg + bin@Wpb + b_p) . wout + b_out
// ---------------------------------------------------------------------------
__global__ void k_final(const float* __restrict__ localf, const float* __restrict__ ctx,
                        const float* __restrict__ binf,
                        const _Float16* __restrict__ Wtpl, const _Float16* __restrict__ Wtpg,
                        const _Float16* __restrict__ Wtpb,
                        const float* __restrict__ biasp, const float* __restrict__ wout,
                        const float* __restrict__ bout, float* __restrict__ scores)
{
  __shared__ __align__(32) _Float16 sLp[16 * HP];
  __shared__ __align__(32) _Float16 sGp[16 * HP];
  __shared__ __align__(32) _Float16 sBin[16 * SKP];
  __shared__ float sdot[16];

  int blk = blockIdx.x;
  int jt = blk % NJT; int t = blk / NJT;
  int i = t % N_; int b = t / N_;
  int j0 = jt * 16;
  int tid = threadIdx.x;

  const float* li = localf + ((size_t)(b * N_ + i)) * HP;
  const float* gi = ctx    + ((size_t)(b * N_ + i)) * HP;
  for (int idx = tid; idx < 16 * HP; idx += 128) {
    int jj = idx / HP, k = idx % HP;
    int j = j0 + jj; int jc = j < N_ ? j : N_ - 1;
    size_t jo = ((size_t)(b * N_ + jc)) * HP + k;
    sLp[idx] = (_Float16)(li[k] + localf[jo]);
    sGp[idx] = (_Float16)(gi[k] + ctx[jo]);
  }
  for (int idx = tid; idx < 16 * SKP; idx += 128) {
    int jj = idx / SKP, k = idx % SKP;
    int j = j0 + jj; int jc = j < N_ ? j : N_ - 1;
    sBin[idx] = (k < BIN_)
      ? (_Float16)binf[((size_t)((b * N_ + i) * N_ + jc)) * BIN_ + k]
      : (_Float16)0.f;
  }
  if (tid < 16) sdot[tid] = bout[0];
  __syncthreads();

  const int lane = tid & 31, wave = tid >> 5;
  const int arow = lane & 15, asel = (lane >> 4) * 16;
  const int bn   = lane & 15, bk   = (lane >> 4) * 16;
  const int cn   = lane & 15, cm   = (lane >> 4) * 8;

  for (int nt = wave; nt < HP / 16; nt += 4) {
    int n0 = nt * 16;
    v8f acc = {};
    {
      const _Float16* Ap = sLp + arow * HP + asel;
      const _Float16* Bp = Wtpl + (size_t)(n0 + bn) * HP + bk;
      for (int kb = 0; kb < HP; kb += 32)
        acc = wmma_step(frag_a(Ap, kb), *(const v16h*)(Bp + kb), acc);
    }
    {
      const _Float16* Ap = sGp + arow * HP + asel;
      const _Float16* Bp = Wtpg + (size_t)(n0 + bn) * HP + bk;
      for (int kb = 0; kb < HP; kb += 32)
        acc = wmma_step(frag_a(Ap, kb), *(const v16h*)(Bp + kb), acc);
    }
    acc = wmma_step(frag_a(sBin + arow * SKP + asel, 0),
                    *(const v16h*)(Wtpb + (size_t)(n0 + bn) * SKP + bk), acc);
    float won = wout[n0 + cn];
    float bvn = biasp[n0 + cn];
#pragma unroll
    for (int r = 0; r < 8; ++r) {
      float v = acc[r] + bvn;
      v = v > 0.f ? v : 0.f;
      atomicAdd(&sdot[cm + r], v * won);
    }
  }
  __syncthreads();
  if (tid < 16) {
    int j = j0 + tid;
    if (j < N_)
      scores[(size_t)(b * N_ + i) * N_ + j] = sdot[tid];
  }
}

// ---------------------------------------------------------------------------
// Per-batch top-K (K=80) by iterative argmax; ties -> lower index (jax order)
// ---------------------------------------------------------------------------
__global__ void k_topk(const float* __restrict__ scores, float* __restrict__ scratch,
                       int* __restrict__ outidx)
{
  constexpr int NT = 256;
  int b = blockIdx.x, tid = threadIdx.x;
  float* sc = scratch + (size_t)b * (N_ * N_);
  for (int idx = tid; idx < N_ * N_; idx += NT)
    sc[idx] = scores[(size_t)b * N_ * N_ + idx];
  __syncthreads();

  __shared__ float sv[NT];
  __shared__ int   si[NT];
  for (int it = 0; it < K_; ++it) {
    float bv = -3.4e38f; int bi = 0x7fffffff;
    for (int idx = tid; idx < N_ * N_; idx += NT) {
      float v = sc[idx];
      if (v > bv) { bv = v; bi = idx; }
    }
    sv[tid] = bv; si[tid] = bi;
    __syncthreads();
    for (int off = NT / 2; off > 0; off >>= 1) {
      if (tid < off) {
        if (sv[tid + off] > sv[tid] ||
            (sv[tid + off] == sv[tid] && si[tid + off] < si[tid])) {
          sv[tid] = sv[tid + off]; si[tid] = si[tid + off];
        }
      }
      __syncthreads();
    }
    if (tid == 0) { outidx[b * K_ + it] = si[0]; sc[si[0]] = -3.4e38f; }
    __syncthreads();
  }
}

// ---------------------------------------------------------------------------
// host orchestration
// ---------------------------------------------------------------------------
extern "C" void kernel_launch(void* const* d_in, const int* in_sizes, int n_in,
                              void* d_out, int out_size, void* d_ws, size_t ws_size,
                              hipStream_t stream)
{
  (void)in_sizes; (void)n_in; (void)out_size; (void)ws_size;

  const float* fatoms   = (const float*)d_in[0];
  const float* fbonds   = (const float*)d_in[1];
  const int*   atom_nb  = (const int*)d_in[2];
  const int*   bond_nb  = (const int*)d_in[3];
  const float* binf     = (const float*)d_in[6];
  const float* mask_nei = (const float*)d_in[7];
  const float* mask_at  = (const float*)d_in[8];
  const float* W_atom      = (const float*)d_in[10];
  const float* W_nei_atom  = (const float*)d_in[11];
  const float* W_nei_bond  = (const float*)d_in[12];
  const float* W_self      = (const float*)d_in[13];
  const float* W_U2        = (const float*)d_in[14];
  const float* b_U2        = (const float*)d_in[15];
  const float* W_U1        = (const float*)d_in[16];
  const float* b_U1        = (const float*)d_in[17];
  const float* W_att_local = (const float*)d_in[18];
  const float* W_att_bin   = (const float*)d_in[19];
  const float* b_att       = (const float*)d_in[20];
  const float* W_att_score = (const float*)d_in[21];
  const float* b_att_score = (const float*)d_in[22];
  const float* W_pl        = (const float*)d_in[23];
  const float* W_pg        = (const float*)d_in[24];
  const float* W_pb        = (const float*)d_in[25];
  const float* b_p         = (const float*)d_in[26];
  const float* W_out       = (const float*)d_in[27];
  const float* b_out       = (const float*)d_in[28];

  char* ws = (char*)d_ws;
  size_t off = 0;
  auto carve = [&](size_t bytes) -> char* {
    char* p = ws + off;
    off = (off + bytes + 255) & ~(size_t)255;
    return p;
  };
  auto h16p = [&](size_t n) { return (_Float16*)carve(n * sizeof(_Float16)); };
  auto f32p = [&](size_t n) { return (float*)carve(n * sizeof(float)); };

  // packed (transposed, f16) weights
  _Float16* Wt_atom  = h16p((size_t)HP * AFP);
  _Float16* Wt_neiA  = h16p((size_t)HP * HP);
  _Float16* Wt_neiB  = h16p((size_t)HP * SKP);
  _Float16* Wt_self  = h16p((size_t)HP * HP);
  _Float16* Wt_U2a   = h16p((size_t)HP * HP);
  _Float16* Wt_U2b   = h16p((size_t)HP * SKP);
  _Float16* Wt_U1    = h16p((size_t)HP * U1KP);
  _Float16* Wt_attl  = h16p((size_t)HP * HP);
  _Float16* Wt_attb  = h16p((size_t)HP * SKP);
  _Float16* Wt_pl    = h16p((size_t)HP * HP);
  _Float16* Wt_pg    = h16p((size_t)HP * HP);
  _Float16* Wt_pb    = h16p((size_t)HP * SKP);
  // padded bias / score vectors (f32)
  float* b_U2p  = f32p(HP);
  float* b_U1p  = f32p(HP);
  float* b_attp = f32p(HP);
  float* b_pp   = f32p(HP);
  float* wscore = f32p(HP);
  float* wout   = f32p(HP);
  // activations
  _Float16* A0     = h16p((size_t)ROWS * AFP);
  _Float16* h16    = h16p((size_t)ROWS * HP);
  _Float16* A_nei  = h16p((size_t)GROWS * HP);
  _Float16* A_bond = h16p((size_t)GROWS * SKP);
  _Float16* A_U1   = h16p((size_t)ROWS * U1KP);
  float* Bnei   = f32p((size_t)GROWS * HP);   // fnei_bond @ W_nei_bond (loop-invariant)
  float* Hnei   = f32p((size_t)GROWS * HP);   // reused as U2out after k_fnei
  float* U2c    = f32p((size_t)GROWS * HP);   // fnei_bond @ W_U2[300:306] (loop-invariant)
  float* fneiB  = f32p((size_t)ROWS * HP);
  float* fselfB = f32p((size_t)ROWS * HP);
  float* localf = f32p((size_t)ROWS * HP);
  float* neilab = f32p((size_t)ROWS * HP);
  float* att_s  = f32p((size_t)NPAIR);
  float* ctx    = f32p((size_t)ROWS * HP);
  float* tscr   = f32p((size_t)B_ * N_ * N_);

  float* scores = (float*)d_out;
  int*   outidx = (int*)d_out + NPAIR;

  auto packwt = [&](const float* W, _Float16* Wt, int K, int N, int Kp, int row0) {
    int total = HP * Kp;
    k_pack_wt<<<(total + 255) / 256, 256, 0, stream>>>(W, Wt, K, N, Kp, HP, N, row0);
  };
  auto packvec = [&](const float* v, float* vp, int N) {
    k_pack_vec<<<(HP + 255) / 256, 256, 0, stream>>>(v, vp, N, HP);
  };
  auto gemm = [&](const _Float16* A, const _Float16* Wt, const float* bias,
                  const float* Cin, float* Cf, _Float16* Ch, int M, int Kp, int relu) {
    dim3 g(M / 16, HP / 64);
    k_gemm<<<g, 128, 0, stream>>>(A, Wt, bias, Cin, Cf, Ch, HP, Kp, relu);
  };

  // ---- weight packing ----
  packwt(W_atom,      Wt_atom, AF_,  H_, AFP,  0);
  packwt(W_nei_atom,  Wt_neiA, H_,   H_, HP,   0);
  packwt(W_nei_bond,  Wt_neiB, BF_,  H_, SKP,  0);
  packwt(W_self,      Wt_self, H_,   H_, HP,   0);
  packwt(W_U2,        Wt_U2a,  H_,   H_, HP,   0);
  packwt(W_U2,        Wt_U2b,  BF_,  H_, SKP,  H_);   // rows 300..305
  packwt(W_U1,        Wt_U1,   2*H_, H_, U1KP, 0);
  packwt(W_att_local, Wt_attl, H_,   H_, HP,   0);
  packwt(W_att_bin,   Wt_attb, BIN_, H_, SKP,  0);
  packwt(W_pl,        Wt_pl,   H_,   H_, HP,   0);
  packwt(W_pg,        Wt_pg,   H_,   H_, HP,   0);
  packwt(W_pb,        Wt_pb,   BIN_, H_, SKP,  0);
  packvec(b_U2, b_U2p, H_);
  packvec(b_U1, b_U1p, H_);
  packvec(b_att, b_attp, H_);
  packvec(b_p, b_pp, H_);
  packvec(W_att_score, wscore, H_);
  packvec(W_out, wout, H_);

  // ---- atom embedding: h = relu(fatoms @ W_atom) ----
  k_pack_fatoms<<<(ROWS * AFP + 255) / 256, 256, 0, stream>>>(fatoms, A0);
  gemm(A0, Wt_atom, nullptr, nullptr, nullptr, h16, ROWS, AFP, 1);

  // ---- loop-invariant bond terms ----
  k_pack_bond<<<(GROWS * SKP + 255) / 256, 256, 0, stream>>>(fbonds, bond_nb, A_bond);
  gemm(A_bond, Wt_neiB, nullptr, nullptr, Bnei, nullptr, GROWS, SKP, 0);
  gemm(A_bond, Wt_U2b,  nullptr, nullptr, U2c,  nullptr, GROWS, SKP, 0);

  // ---- message-passing depth loop ----
  for (int t = 0; t < 3; ++t) {
    k_gather_nei<<<(GROWS * HP + 255) / 256, 256, 0, stream>>>(h16, atom_nb, A_nei);
    gemm(A_nei, Wt_neiA, nullptr, nullptr, Hnei, nullptr, GROWS, HP, 0);
    k_fnei<<<ROWS, HP, 0, stream>>>(Hnei, Bnei, mask_nei, fneiB);
    gemm(h16, Wt_self, nullptr, nullptr, fselfB, nullptr, ROWS, HP, 0);
    k_local<<<ROWS, HP, 0, stream>>>(fneiB, fselfB, mask_at, localf);
    // nei_label = relu(concat(fnei_atom, fnei_bond) @ W_U2 + b_U2)
    gemm(A_nei, Wt_U2a, b_U2p, U2c, Hnei /* U2out reuse */, nullptr, GROWS, HP, 1);
    k_labelsum<<<ROWS, HP, 0, stream>>>(Hnei, mask_nei, neilab);
    k_pack_u1<<<(ROWS * U1KP + 255) / 256, 256, 0, stream>>>(h16, neilab, A_U1);
    gemm(A_U1, Wt_U1, b_U1p, nullptr, nullptr, h16, ROWS, U1KP, 1);
  }

  // ---- fused pair attention + context + final head ----
  k_att<<<B_ * N_ * NJT, 128, 0, stream>>>(localf, binf, Wt_attl, Wt_attb,
                                           b_attp, wscore, b_att_score, att_s);
  k_ctx<<<ROWS, HP, 0, stream>>>(att_s, localf, ctx);
  k_final<<<B_ * N_ * NJT, 128, 0, stream>>>(localf, ctx, binf, Wt_pl, Wt_pg,
                                             Wt_pb, b_pp, wout, b_out, scores);

  // ---- top-K per batch ----
  k_topk<<<B_, 256, 0, stream>>>(scores, tscr, outidx);
}
